// DCGRUDecoder_86285892976922
// MI455X (gfx1250) — matrix operand order repaired and checked by reading, not compiled, not run on hardware
//
#include <hip/hip_runtime.h>
#include <hip/hip_bf16.h>

// ---------------------------------------------------------------------------
// DCGRU decoder for MI455X (gfx1250), f16 WMMA with f32 accumulate.
// Layouts: activations [n(1024)][b(32)][c] ("row" = n*32+b), supports [s][n][m].
// Support GEMM: ping-pong double-buffered K loop (no register rotation copies).
// Weight contraction: compile-time shapes, fully unrolled K loop with
// per-k-step grouped loads (2 vector A + 64 scalar B loads -> 4 back-to-back WMMAs).
// ---------------------------------------------------------------------------

typedef __attribute__((ext_vector_type(16))) _Float16 v16h;
typedef __attribute__((ext_vector_type(8)))  _Float16 v8h;
typedef __attribute__((ext_vector_type(8)))  float    v8f;

#define NN   1024
#define BB   32
#define HH   64
#define NB   (NN*BB)       // 32768 rows

// ---- WMMA fragment loaders (CDNA5 16x16x32 f16 layouts, wave32) -----------

// A fragment: 16(M) x 32(K), A row-major with leading dim lda (halfs).
// lanes 0-15: M=lane, K chunks [0..7] and [16..23]; lanes 16-31: M=lane-16,
// K chunks [8..15] and [24..31].
__device__ __forceinline__ v16h ldA(const _Float16* __restrict__ A, int lda, int lane) {
  const _Float16* p = A + (lane & 15) * lda + ((lane >> 4) << 3);
  union { v16h v; v8h h[2]; } u;
  u.h[0] = *(const v8h*)(p);
  u.h[1] = *(const v8h*)(p + 16);
  return u.v;
}

// B fragment: 32(K) x 16(N), B row-major with leading dim ldb (halfs).
// lanes 0-15: N=lane, K [0..7] and [16..23]; lanes 16-31: N=lane-16,
// K [8..15] and [24..31].
__device__ __forceinline__ v16h ldB(const _Float16* __restrict__ B, int ldb, int lane) {
  const _Float16* p = B + (lane & 15) + ((lane >> 4) << 3) * ldb;
  v16h v;
#pragma unroll
  for (int i = 0; i < 8; ++i) { v[i] = p[i * ldb]; v[8 + i] = p[(16 + i) * ldb]; }
  return v;
}

#define WMMA(a, b, c) __builtin_amdgcn_wmma_f32_16x16x32_f16( \
    false, (a), false, (b), (short)0, (c), false, false)

// ---- support GEMM: OUT[n,col] = f16( alpha * S@X + beta * D ) -------------
// S: [1024,1024] f16.  X, D, OUT: [1024, NCOLS] f16.  Wave tile: 16 x 64.
template<int NCOLS>
__global__ __launch_bounds__(128) void support_gemm_k(
    const _Float16* __restrict__ S, const _Float16* __restrict__ X,
    const _Float16* __restrict__ D, _Float16* __restrict__ OUT,
    float alpha, float beta)
{
  const int lane  = threadIdx.x & 31;
  const int wave  = blockIdx.x * (blockDim.x >> 5) + (threadIdx.x >> 5);
  constexpr int wavesN = NCOLS / 64;
  const int m0 = (wave / wavesN) << 4;
  const int n0 = (wave % wavesN) << 6;
  const _Float16* Sa = S + m0 * NN;
  const _Float16* Xb = X + n0;

  v8f acc[4] = {};
  v16h a0, a1, b0[4], b1[4];

  // prologue: k-sets 0 and 32
  a0 = ldA(Sa + 0, NN, lane);
#pragma unroll
  for (int j = 0; j < 4; ++j) b0[j] = ldB(Xb + 0 * NCOLS + 16 * j, NCOLS, lane);
  a1 = ldA(Sa + 32, NN, lane);
#pragma unroll
  for (int j = 0; j < 4; ++j) b1[j] = ldB(Xb + 32 * NCOLS + 16 * j, NCOLS, lane);

  // steady state: consume a set, immediately refill it (ping-pong, no copies)
  for (int k0 = 64; k0 <= NN - 64; k0 += 64) {
#pragma unroll
    for (int j = 0; j < 4; ++j) acc[j] = WMMA(a0, b0[j], acc[j]);
    a0 = ldA(Sa + k0, NN, lane);
#pragma unroll
    for (int j = 0; j < 4; ++j) b0[j] = ldB(Xb + k0 * NCOLS + 16 * j, NCOLS, lane);
#pragma unroll
    for (int j = 0; j < 4; ++j) acc[j] = WMMA(a1, b1[j], acc[j]);
    a1 = ldA(Sa + (k0 + 32), NN, lane);
#pragma unroll
    for (int j = 0; j < 4; ++j) b1[j] = ldB(Xb + (k0 + 32) * NCOLS + 16 * j, NCOLS, lane);
  }
  // tail: last two sets
#pragma unroll
  for (int j = 0; j < 4; ++j) acc[j] = WMMA(a0, b0[j], acc[j]);
#pragma unroll
  for (int j = 0; j < 4; ++j) acc[j] = WMMA(a1, b1[j], acc[j]);

  const int mrow = m0 + ((lane >> 4) << 3);
  const int col0 = n0 + (lane & 15);
#pragma unroll
  for (int j = 0; j < 4; ++j) {
#pragma unroll
    for (int r = 0; r < 8; ++r) {
      int idx = (mrow + r) * NCOLS + col0 + 16 * j;
      float v = alpha * acc[j][r];
      if (beta != 0.0f) v += beta * (float)D[idx];
      OUT[idx] = (_Float16)v;
    }
  }
}

// ---- weight contraction: ACC[row,o] (+)= A @ W  (f32 accumulator) ---------
// A: [NB, Cp] f16.  W: [Cp, O] f16.  initMode: 1 -> ACC = bias + A@W.
template<int Cp, int O>
__global__ __launch_bounds__(128) void wcontract_k(
    const _Float16* __restrict__ A, const _Float16* __restrict__ W,
    const float* __restrict__ bias, float* __restrict__ ACC, int initMode)
{
  const int lane  = threadIdx.x & 31;
  const int wave  = blockIdx.x * (blockDim.x >> 5) + (threadIdx.x >> 5);
  constexpr int wavesN = O / 64;
  const int m0 = (wave / wavesN) << 4;
  const int n0 = (wave % wavesN) << 6;
  const int mrow = m0 + ((lane >> 4) << 3);
  const int nlane = lane & 15;

  v8f acc[4];
#pragma unroll
  for (int j = 0; j < 4; ++j) {
    if (initMode) {
      float bv = bias[n0 + 16 * j + nlane];
#pragma unroll
      for (int r = 0; r < 8; ++r) acc[j][r] = bv;
    } else {
#pragma unroll
      for (int r = 0; r < 8; ++r)
        acc[j][r] = ACC[(mrow + r) * O + n0 + 16 * j + nlane];
    }
  }

  // fully unrolled; loads for each k-step grouped ahead of its 4 WMMAs so the
  // memory clause overlaps the preceding step's matrix ops.
#pragma unroll
  for (int k0 = 0; k0 < Cp; k0 += 32) {
    v16h a = ldA(A + m0 * Cp + k0, Cp, lane);
    v16h b[4];
#pragma unroll
    for (int j = 0; j < 4; ++j) b[j] = ldB(W + k0 * O + n0 + 16 * j, O, lane);
#pragma unroll
    for (int j = 0; j < 4; ++j) acc[j] = WMMA(a, b[j], acc[j]);
  }

#pragma unroll
  for (int j = 0; j < 4; ++j)
#pragma unroll
    for (int r = 0; r < 8; ++r)
      ACC[(mrow + r) * O + n0 + 16 * j + nlane] = acc[j][r];
}

// ---- setup / elementwise kernels ------------------------------------------

__global__ void f32_to_f16_k(const float* __restrict__ in, _Float16* __restrict__ out, int n) {
  int i = blockIdx.x * blockDim.x + threadIdx.x;
  if (i < n) out[i] = (_Float16)in[i];
}

// W[m][C][O] f32 -> Wp[m][Cp][O] f16, zero-padded in C.
__global__ void pack_w_k(const float* __restrict__ W, _Float16* __restrict__ Wp,
                         int M, int C, int Cp, int O) {
  int i = blockIdx.x * blockDim.x + threadIdx.x;
  int total = M * Cp * O;
  if (i >= total) return;
  int o = i % O; int t = i / O; int c = t % Cp; int m = t / Cp;
  Wp[i] = (c < C) ? (_Float16)W[(m * C + c) * O + o] : (_Float16)0.0f;
}

// encoder_state [2][B][N][H] -> state[l][n*B+b][H]
__global__ void init_state_k(const float* __restrict__ enc,
                             float* __restrict__ s0, float* __restrict__ s1) {
  int i = blockIdx.x * blockDim.x + threadIdx.x;
  int total = BB * NN * HH;
  if (i >= total) return;
  int c = i % HH; int t = i / HH; int n = t % NN; int b = t / NN;
  int dst = (n * BB + b) * HH + c;
  s0[dst] = enc[i];
  s1[dst] = enc[total + i];
}

__global__ void zero_f32_k(float* __restrict__ p, int n) {
  int i = blockIdx.x * blockDim.x + threadIdx.x;
  if (i < n) p[i] = 0.0f;
}

// layer0 concat: [0:2]=dec_in, [2:66]=state0 (optionally * r), [66:96]=0
__global__ void build_xcat0_k(const float* __restrict__ dec_in,
                              const float* __restrict__ state,
                              const float* __restrict__ gate,
                              _Float16* __restrict__ xc, int useR) {
  int i = blockIdx.x * blockDim.x + threadIdx.x;
  int total = NB * 96;
  if (i >= total) return;
  int cp = i % 96; int row = i / 96;
  float v;
  if (cp < 2) v = dec_in[row * 2 + cp];
  else if (cp < 66) {
    v = state[row * HH + (cp - 2)];
    if (useR) v *= gate[row * 128 + (cp - 2)];
  } else v = 0.0f;
  xc[i] = (_Float16)v;
}

// layer1 concat: [0:64]=state0, [64:128]=state1 (optionally * r)
__global__ void build_xcat1_k(const float* __restrict__ s0,
                              const float* __restrict__ s1,
                              const float* __restrict__ gate,
                              _Float16* __restrict__ xc, int useR) {
  int i = blockIdx.x * blockDim.x + threadIdx.x;
  int total = NB * 128;
  if (i >= total) return;
  int cp = i % 128; int row = i / 128;
  float v;
  if (cp < 64) v = s0[row * HH + cp];
  else {
    v = s1[row * HH + (cp - 64)];
    if (useR) v *= gate[row * 128 + (cp - 64)];
  }
  xc[i] = (_Float16)v;
}

__global__ void sigmoid_k(float* __restrict__ g, int n) {
  int i = blockIdx.x * blockDim.x + threadIdx.x;
  if (i < n) g[i] = 1.0f / (1.0f + expf(-g[i]));
}

// state = u*state + (1-u)*tanh(hpre);  u = gate[row*128 + 64 + c]
__global__ void gru_update_k(float* __restrict__ state, const float* __restrict__ gate,
                             const float* __restrict__ hpre, int n) {
  int i = blockIdx.x * blockDim.x + threadIdx.x;
  if (i >= n) return;
  int c = i % HH; int row = i / HH;
  float u = gate[row * 128 + 64 + c];
  float h = tanhf(hpre[i]);
  state[i] = u * state[i] + (1.0f - u) * h;
}

// out[t,b,n,o] = state1[n,b,:] @ W_p + b_p ; also dec_in = out (in [n,b,o])
__global__ void project_k(const float* __restrict__ s1, const float* __restrict__ Wp,
                          const float* __restrict__ bp, float* __restrict__ out_t,
                          float* __restrict__ dec_in) {
  int i = blockIdx.x * blockDim.x + threadIdx.x;
  int total = NB * 2;
  if (i >= total) return;
  int o = i & 1; int row = i >> 1;
  const float* s = s1 + row * HH;
  float acc = bp[o];
#pragma unroll
  for (int c = 0; c < HH; ++c) acc += s[c] * Wp[c * 2 + o];
  dec_in[i] = acc;
  int n = row >> 5; int b = row & 31;
  out_t[(b * NN + n) * 2 + o] = acc;
}

// ---------------------------------------------------------------------------

template<int Cp, int O>
static void run_cheb(const _Float16* S16, _Float16* xcat, _Float16* t1b, _Float16* t2b,
                     const _Float16* W16, const float* bias, float* ACC,
                     hipStream_t stream) {
  constexpr int ncols = BB * Cp;
  constexpr int supBlocks = (64 * (ncols / 64)) / 4;   // waves/4, blockDim=128
  constexpr int wcBlocks  = ((NB / 16) * (O / 64)) / 4;
  constexpr int Wstride = Cp * O;
  // m = 0 : identity feature (init with bias)
  wcontract_k<Cp, O><<<wcBlocks, 128, 0, stream>>>(xcat, W16, bias, ACC, 1);
  for (int s = 0; s < 2; ++s) {
    const _Float16* Ss = S16 + (size_t)s * NN * NN;
    // T1 = S @ x
    support_gemm_k<ncols><<<supBlocks, 128, 0, stream>>>(Ss, xcat, xcat, t1b, 1.0f, 0.0f);
    wcontract_k<Cp, O><<<wcBlocks, 128, 0, stream>>>(t1b, W16 + (1 + 2 * s) * Wstride, bias, ACC, 0);
    // T2 = 2*S @ T1 - x
    support_gemm_k<ncols><<<supBlocks, 128, 0, stream>>>(Ss, t1b, xcat, t2b, 2.0f, -1.0f);
    wcontract_k<Cp, O><<<wcBlocks, 128, 0, stream>>>(t2b, W16 + (2 + 2 * s) * Wstride, bias, ACC, 0);
  }
}

extern "C" void kernel_launch(void* const* d_in, const int* in_sizes, int n_in,
                              void* d_out, int out_size, void* d_ws, size_t ws_size,
                              hipStream_t stream) {
  const float* enc  = (const float*)d_in[0];
  const float* sup  = (const float*)d_in[1];
  const float* Wru0 = (const float*)d_in[2];
  const float* bru0 = (const float*)d_in[3];
  const float* Wh0  = (const float*)d_in[4];
  const float* bh0  = (const float*)d_in[5];
  const float* Wru1 = (const float*)d_in[6];
  const float* bru1 = (const float*)d_in[7];
  const float* Wh1  = (const float*)d_in[8];
  const float* bh1  = (const float*)d_in[9];
  const float* Wp   = (const float*)d_in[10];
  const float* bp   = (const float*)d_in[11];
  float* out = (float*)d_out;

  // workspace carve-out (~72.5 MB total; fits comfortably in 192 MB L2)
  char* wsb = (char*)d_ws;
  size_t off = 0;
  auto carve = [&](size_t bytes) -> char* {
    char* p = wsb + off;
    off = (off + bytes + 255) & ~size_t(255);
    return p;
  };
  _Float16* S16     = (_Float16*)carve(size_t(2) * NN * NN * 2);
  _Float16* Wru0_16 = (_Float16*)carve(size_t(5) * 96 * 128 * 2);
  _Float16* Wh0_16  = (_Float16*)carve(size_t(5) * 96 * 64 * 2);
  _Float16* Wru1_16 = (_Float16*)carve(size_t(5) * 128 * 128 * 2);
  _Float16* Wh1_16  = (_Float16*)carve(size_t(5) * 128 * 64 * 2);
  float*    state0  = (float*)   carve(size_t(NB) * HH * 4);
  float*    state1  = (float*)   carve(size_t(NB) * HH * 4);
  float*    dec_in  = (float*)   carve(size_t(NB) * 2 * 4);
  _Float16* xcat    = (_Float16*)carve(size_t(NB) * 128 * 2);
  _Float16* t1b     = (_Float16*)carve(size_t(NB) * 128 * 2);
  _Float16* t2b     = (_Float16*)carve(size_t(NB) * 128 * 2);
  float*    gate    = (float*)   carve(size_t(NB) * 128 * 4);
  float*    hpre    = (float*)   carve(size_t(NB) * HH * 4);

  auto g256 = [](int n) { return (n + 255) / 256; };

  // ---- one-time-per-call setup (deterministic, re-done every call) ----
  f32_to_f16_k<<<g256(2 * NN * NN), 256, 0, stream>>>(sup, S16, 2 * NN * NN);
  pack_w_k<<<g256(5 * 96 * 128), 256, 0, stream>>>(Wru0, Wru0_16, 5, 66, 96, 128);
  pack_w_k<<<g256(5 * 96 * 64),  256, 0, stream>>>(Wh0,  Wh0_16,  5, 66, 96, 64);
  pack_w_k<<<g256(5 * 128 * 128),256, 0, stream>>>(Wru1, Wru1_16, 5, 128, 128, 128);
  pack_w_k<<<g256(5 * 128 * 64), 256, 0, stream>>>(Wh1,  Wh1_16,  5, 128, 128, 64);
  init_state_k<<<g256(BB * NN * HH), 256, 0, stream>>>(enc, state0, state1);
  zero_f32_k<<<g256(NB * 2), 256, 0, stream>>>(dec_in, NB * 2);

  for (int t = 0; t < 12; ++t) {
    // ---- layer 0 ----
    build_xcat0_k<<<g256(NB * 96), 256, 0, stream>>>(dec_in, state0, gate, xcat, 0);
    run_cheb<96, 128>(S16, xcat, t1b, t2b, Wru0_16, bru0, gate, stream);
    sigmoid_k<<<g256(NB * 128), 256, 0, stream>>>(gate, NB * 128);
    build_xcat0_k<<<g256(NB * 96), 256, 0, stream>>>(dec_in, state0, gate, xcat, 1);
    run_cheb<96, 64>(S16, xcat, t1b, t2b, Wh0_16, bh0, hpre, stream);
    gru_update_k<<<g256(NB * HH), 256, 0, stream>>>(state0, gate, hpre, NB * HH);
    // ---- layer 1 ----
    build_xcat1_k<<<g256(NB * 128), 256, 0, stream>>>(state0, state1, gate, xcat, 0);
    run_cheb<128, 128>(S16, xcat, t1b, t2b, Wru1_16, bru1, gate, stream);
    sigmoid_k<<<g256(NB * 128), 256, 0, stream>>>(gate, NB * 128);
    build_xcat1_k<<<g256(NB * 128), 256, 0, stream>>>(state0, state1, gate, xcat, 1);
    run_cheb<128, 64>(S16, xcat, t1b, t2b, Wh1_16, bh1, hpre, stream);
    gru_update_k<<<g256(NB * HH), 256, 0, stream>>>(state1, gate, hpre, NB * HH);
    // ---- projection ----
    project_k<<<g256(NB * 2), 256, 0, stream>>>(state1, Wp, bp, out + (size_t)t * BB * NN * 2, dec_in);
  }
  (void)in_sizes; (void)n_in; (void)out_size; (void)ws_size;
}
